// SS2D_43963285242154
// MI455X (gfx1250) — compile-verified, matrix operand order checked by reading, not run on hardware
//
#include <hip/hip_runtime.h>
#include <math.h>
#include <stdint.h>

// ---------------------------------------------------------------------------
// SS2D (VMamba selective scan 2D) for gfx1250.
// B=4, H=W=64, L=4096, D_MODEL=96, D_INNER=192, D_STATE=16, DT_RANK=6.
// GEMMs: v_wmma_f32_16x16x32_f16 with pre-swizzled f16 weight fragments
// (global_load_b128 per fragment) and LDS-swizzled A fragments (ds_load_b128).
// Scan: async global->LDS double-buffered B/C lines (ASYNCcnt) to hide the
// broadcast-load latency behind the 16 v_exp_f32 of each serial step.
// ---------------------------------------------------------------------------

typedef _Float16 f16;
typedef __attribute__((ext_vector_type(16))) _Float16 v16h;
typedef __attribute__((ext_vector_type(8)))  float    v8f;

#define L_    4096
#define DM_   96
#define DI_   192
#define NS_   16
#define NX_   38    // dt_rank(6) + 2*16
#define NZ_   384   // 2*DI_

// ---------------------------------------------------------------------------
// Fragment swizzle (CDNA5 wave32 16-bit A/B layout for 16x16x32):
//   element (m_or_n = lane&15, k = ks*32 + (g<<4) + (hf<<3) + (vv<<1) + e)
//   lives at lane = (m|n) | (hf<<4), slot j = (g<<3) | (vv<<1) | e.
// Packed: frag[lane][j] -> 16 contiguous f16 per lane -> one v16h (32B) load.
// Note a_sw(r,c) = a_sw(0,c) + 16*r  (r enters only via the lane low bits).
// ---------------------------------------------------------------------------
__device__ __forceinline__ int a_sw(int r, int c) {        // swizzled LDS index
  const int ks = c >> 5, kl = c & 31;
  const int ln = (r & 15) | ((kl & 8) << 1);
  const int j  = ((kl >> 4) << 3) | (kl & 7);
  return ((ks << 5) + ln) * 16 + j;
}

// direction permutation: row l of direction k maps to spatial position pos.
// Within a 16-aligned tile all four are affine with these (pos0, step):
__device__ __forceinline__ void dir_affine(int k, int l0, int& pos0, int& step) {
  if      (k == 0) { pos0 = l0;            step = 1;   }
  else if (k == 1) { pos0 = (L_ - 1) - l0; step = -1;  }
  else if (k == 2) { pos0 = ((l0 & 63) << 6) | (l0 >> 6); step = 64; }
  else { const int lp = (L_ - 1) - l0; pos0 = ((lp & 63) << 6) | (lp >> 6); step = -64; }
}

// ---- weight pre-pack: W[K][N] (fp32) -> fragment-ordered f16, zero-padded --
__global__ void __launch_bounds__(256) k_pack(const float* __restrict__ W,
                                              f16* __restrict__ Wp,
                                              int K, int N, int nks, int total) {
  const int idx = blockIdx.x * 256 + threadIdx.x;
  if (idx >= total) return;
  const int j    = idx & 15;
  const int lane = (idx >> 4) & 31;
  const int rest = idx >> 9;
  const int ks   = rest % nks;
  const int nt   = rest / nks;
  const int e = j & 1, v = j >> 1, g = v >> 2, vv = v & 3, hf = lane >> 4;
  const int k = ks * 32 + (g << 4) + (hf << 3) + (vv << 1) + e;
  const int n = (nt << 4) + (lane & 15);
  Wp[idx] = (k < K && n < N) ? (f16)W[k * N + n] : (f16)0.f;
}

__device__ __forceinline__ v16h b_frag(const f16* __restrict__ Wp,
                                       int nks, int nt, int ks, int lane) {
  return *(const v16h*)(Wp + (((nt * nks + ks) << 5) + lane) * 16);
}
__device__ __forceinline__ v16h a_frag(const f16* At, int ks, int lane) {
  return *(const v16h*)(At + ((ks << 5) + lane) * 16);
}

// ---- Kernel 1: xz = x @ W_in (single output buffer, branch-free stores) ---
// M = B*L = 16384 (16-row tiles), K = 96 (3 ks), N = 384 (24 n-tiles).
__global__ void __launch_bounds__(256) k_inproj(const float* __restrict__ x,
                                                const f16* __restrict__ WinP,
                                                float* __restrict__ xz) {
  __shared__ __align__(32) f16 At[3 * 512];
  const int R0 = blockIdx.x << 4;
  {
    const int row = threadIdx.x >> 4, c0 = threadIdx.x & 15;
    const float* xr = x + (size_t)(R0 + row) * DM_;
#pragma unroll
    for (int i = 0; i < 6; ++i) {
      const int c = c0 + (i << 4);
      At[a_sw(row, c)] = (f16)xr[c];
    }
  }
  __syncthreads();

  const int lane = threadIdx.x & 31;
  const int wave = threadIdx.x >> 5;
#pragma unroll
  for (int j = 0; j < 3; ++j) {
    const int nt = wave * 3 + j;
    v8f acc = {};
#pragma unroll
    for (int ks = 0; ks < 3; ++ks) {
      acc = __builtin_amdgcn_wmma_f32_16x16x32_f16(false, a_frag(At, ks, lane),
                                                   false, b_frag(WinP, 3, nt, ks, lane),
                                                   (short)0, acc, false, false);
    }
    const int n = (nt << 4) + (lane & 15), hf = lane >> 4;
    float* p = xz + (size_t)(R0 + (hf << 3)) * NZ_ + n;
#pragma unroll
    for (int r = 0; r < 8; ++r) p[r * NZ_] = acc[r];
  }
}

// ---- Kernel 2: depthwise 3x3 conv (SAME) + bias + SiLU --------------------
__global__ void __launch_bounds__(DI_) k_conv(const float* __restrict__ xz,
                                              const float* __restrict__ cw,
                                              const float* __restrict__ cb,
                                              float* __restrict__ xo) {
  const int d = threadIdx.x;
  const int p = blockIdx.x;                       // b*4096 + h*64 + w
  const int w = p & 63, h = (p >> 6) & 63, b = p >> 12;
  float acc = cb[d];
#pragma unroll
  for (int kh = 0; kh < 3; ++kh) {
    const int ih = h + kh - 1;
    if (ih < 0 || ih > 63) continue;
#pragma unroll
    for (int kw = 0; kw < 3; ++kw) {
      const int iw = w + kw - 1;
      if (iw < 0 || iw > 63) continue;
      acc += xz[(size_t)(((b << 6) + ih) * 64 + iw) * NZ_ + d] *
             cw[(kh * 3 + kw) * DI_ + d];
    }
  }
  xo[(size_t)p * DI_ + d] = acc / (1.f + __expf(-acc));     // SiLU
}

// ---- Kernel 3: x_dbl = xs @ W_xproj ; delta = softplus(dt@W_dt + b) -------
// Rows R = ((b*4+k)*L + l), 16 rows/block, 192 threads (6 waves; 3 do WMMA).
__global__ void __launch_bounds__(DI_) k_xproj(const float* __restrict__ xconv,
                                               const f16* __restrict__ WxpP,
                                               const float* __restrict__ Wdt,
                                               const float* __restrict__ bdt,
                                               float* __restrict__ delta,
                                               float* __restrict__ BC) {
  __shared__ __align__(32) f16 At[6 * 512];
  __shared__ float xdbl[16][48];
  const int R0 = blockIdx.x << 4;
  const int bk = R0 >> 12, b = bk >> 2, k = bk & 3;
  const int d  = threadIdx.x;                     // 0..191

  // stage A tile: affine permuted rows (one-time direction decode)
  {
    int pos0, step;
    dir_affine(k, R0 & (L_ - 1), pos0, step);
    const float* src = xconv + (size_t)(b * L_ + pos0) * DI_ + d;
    const int sstep = step * DI_;
    f16* ap = At + a_sw(0, d);                    // +16 per row
#pragma unroll
    for (int i = 0; i < 16; ++i) {
      ap[i << 4] = (f16)(*src);
      src += sstep;
    }
  }
  __syncthreads();

  const int lane = threadIdx.x & 31;
  const int wave = threadIdx.x >> 5;
  if (wave < 3) {                                 // 3 n-tiles cover N=38 (padded)
    v8f acc = {};
#pragma unroll
    for (int ks = 0; ks < 6; ++ks) {
      acc = __builtin_amdgcn_wmma_f32_16x16x32_f16(false, a_frag(At, ks, lane),
                                                   false, b_frag(WxpP, 6, wave, ks, lane),
                                                   (short)0, acc, false, false);
    }
    const int n = (wave << 4) + (lane & 15), hf = lane >> 4;
#pragma unroll
    for (int r = 0; r < 8; ++r) xdbl[r + (hf << 3)][n] = acc[r];
  }
  __syncthreads();

  // delta = softplus(dt @ W_dt + b_dt): weights hoisted per thread
  {
    float wdt[6];
#pragma unroll
    for (int j = 0; j < 6; ++j) wdt[j] = Wdt[j * DI_ + d];
    const float bd = bdt[d];
    float* dp = delta + (size_t)R0 * DI_ + d;
#pragma unroll
    for (int r = 0; r < 16; ++r) {
      float t = bd;
#pragma unroll
      for (int j = 0; j < 6; ++j) t += xdbl[r][j] * wdt[j];
      dp[r * DI_] = (t > 20.f) ? t : log1pf(__expf(t));
    }
  }
  // B (cols 0..15) and C (cols 16..31) -> unified BC buffer, branch-free
  for (int e = threadIdx.x; e < 16 * 32; e += DI_) {
    const int r = e >> 5, c = e & 31;
    BC[((size_t)(R0 + r) << 5) + c] = xdbl[r][6 + c];
  }
}

// ---- Kernel 4: selective scan (fused, y overwrites delta in place) --------
// One wave/block. B|C line for step l+1 is copied global->LDS by the async
// engine (one GLOBAL_LOAD_ASYNC_TO_LDS_B32, 32 lanes x 4B = 128B) while the
// VALU runs step l; consumption gated by s_wait_asynccnt 1 (in-order).
__global__ void __launch_bounds__(32) k_scan(const float* __restrict__ xconv,
                                             const float* __restrict__ Alogs,
                                             float* __restrict__ dy,   // delta in / y out
                                             const float* __restrict__ BC) {
  __shared__ __align__(16) float bcb[2][32];      // double-buffered B|C line
  const int bk = blockIdx.x / 6, chunk = blockIdx.x % 6;
  const int b = bk >> 2, k = bk & 3;
  const int tid = threadIdx.x;
  const int d = (chunk << 5) + tid;

  float a[NS_], h[NS_];
#pragma unroll
  for (int n = 0; n < NS_; ++n) {
    a[n] = -__expf(Alogs[((size_t)(k * DI_ + d)) * NS_ + n]);
    h[n] = 0.f;
  }
  const float* gbc = BC + ((size_t)bk * L_ << 5) + tid;   // lane's dword in line
  float*       dyp = dy + (size_t)bk * L_ * DI_;
  const float* xp  = xconv + (size_t)b * L_ * DI_ + d;

  const uint32_t lds0 = (uint32_t)(uintptr_t)&bcb[0][tid];
  const uint32_t lds1 = (uint32_t)(uintptr_t)&bcb[1][tid];

  int pos, step;
  dir_affine(k, 0, pos, step);                    // wrap-corrected affine walk

  // prologue: async-copy line 0 into buffer 0 (ASYNCcnt -> 1)
  asm volatile("global_load_async_to_lds_b32 %0, %1, off"
               :: "v"(lds0), "v"(gbc) : "memory");

  for (int l = 0; l < L_; ++l) {
    // issue next line (wraps to line 0 on the last step to keep cnt invariant)
    const int lnext = (l + 1) & (L_ - 1);
    const uint32_t ldsn = ((l + 1) & 1) ? lds1 : lds0;
    asm volatile("global_load_async_to_lds_b32 %0, %1, off"
                 :: "v"(ldsn), "v"(gbc + ((size_t)lnext << 5)) : "memory");

    const int idx = l * DI_ + d;
    const float delta = dyp[idx];
    const float xv    = xp[pos * DI_];
    if (l + 8 < L_) __builtin_prefetch(&dyp[idx + 8 * DI_], 0, 1); // global_prefetch_b8

    pos += step;                                  // branchless wrap (transpose dirs)
    if (pos >= L_) pos -= (L_ - 1);
    if (pos < 0)   pos += (L_ - 1);

    // line l complete when <=1 async op outstanding (in-order completion)
    asm volatile("s_wait_asynccnt 0x1" ::: "memory");
    const float du = delta * xv;
    const float4* Q = (const float4*)&bcb[l & 1][0];     // 128B: B[16] | C[16]
    const float4 Bq0 = Q[0], Bq1 = Q[1], Bq2 = Q[2], Bq3 = Q[3];
    const float4 Cq0 = Q[4], Cq1 = Q[5], Cq2 = Q[6], Cq3 = Q[7];
    const float bv[NS_] = {Bq0.x, Bq0.y, Bq0.z, Bq0.w, Bq1.x, Bq1.y, Bq1.z, Bq1.w,
                           Bq2.x, Bq2.y, Bq2.z, Bq2.w, Bq3.x, Bq3.y, Bq3.z, Bq3.w};
    const float cv[NS_] = {Cq0.x, Cq0.y, Cq0.z, Cq0.w, Cq1.x, Cq1.y, Cq1.z, Cq1.w,
                           Cq2.x, Cq2.y, Cq2.z, Cq2.w, Cq3.x, Cq3.y, Cq3.z, Cq3.w};
    float y = 0.f;
#pragma unroll
    for (int n = 0; n < NS_; ++n) {
      const float dA = __expf(delta * a[n]);   // v_exp_f32
      h[n] = dA * h[n] + du * bv[n];
      y += h[n] * cv[n];
    }
    dyp[idx] = y;                              // overwrite delta (already consumed)
  }
}

// ---- Kernel 5: merge 4 directions + skip + SiLU gate, then @ W_out --------
// M-tile 16 rows, K = 192 (6 ks), N = 96 (6 waves, one n-tile each).
__global__ void __launch_bounds__(DI_) k_out(const float* __restrict__ y,
                                             const float* __restrict__ xconv,
                                             const float* __restrict__ xz,
                                             const float* __restrict__ Ds,
                                             const f16* __restrict__ WoutP,
                                             float* __restrict__ out) {
  __shared__ __align__(32) f16 Gt[6 * 512];
  const int R0 = blockIdx.x << 4;               // row in [0, B*L)
  const int b = R0 >> 12;
  const int l0 = R0 & (L_ - 1);
  const int d = threadIdx.x;                    // 0..191

  {
    // four incremental gather pointers (affine within the 16-row tile)
    const int lp0 = (L_ - 1) - l0;
    const int lt0  = ((l0  & 63) << 6) | (l0  >> 6);
    const int lt0r = ((lp0 & 63) << 6) | (lp0 >> 6);
    const float* yb = y + (size_t)b * 4 * L_ * DI_ + d;
    const float* p0 = yb + (size_t)(0 * L_ + l0)   * DI_;   // step +DI_
    const float* p1 = yb + (size_t)(1 * L_ + lp0)  * DI_;   // step -DI_
    const float* p2 = yb + (size_t)(2 * L_ + lt0)  * DI_;   // step +64*DI_
    const float* p3 = yb + (size_t)(3 * L_ + lt0r) * DI_;   // step -64*DI_
    const float* pxv = xconv + (size_t)R0 * DI_ + d;         // step +DI_
    const float* pzv = xz + (size_t)R0 * NZ_ + DI_ + d;      // step +NZ_
    const float sd = Ds[d] + Ds[DI_ + d] + Ds[2 * DI_ + d] + Ds[3 * DI_ + d];
    f16* gp = Gt + a_sw(0, d);                               // +16 per row
#pragma unroll
    for (int r = 0; r < 16; ++r) {
      const float ys = p0[r * DI_] + p1[-r * DI_] + p2[r * 64 * DI_] + p3[-r * 64 * DI_];
      const float xv = pxv[r * DI_];
      const float zv = pzv[r * NZ_];
      gp[r << 4] = (f16)((ys + xv * sd) * (zv / (1.f + __expf(-zv))));
    }
  }
  __syncthreads();

  const int lane = threadIdx.x & 31;
  const int wave = threadIdx.x >> 5;            // 0..5 -> n-tiles over N=96
  v8f acc = {};
#pragma unroll
  for (int ks = 0; ks < 6; ++ks) {
    acc = __builtin_amdgcn_wmma_f32_16x16x32_f16(false, a_frag(Gt, ks, lane),
                                                 false, b_frag(WoutP, 6, wave, ks, lane),
                                                 (short)0, acc, false, false);
  }
  const int n = (wave << 4) + (lane & 15), hf = lane >> 4;
  float* p = out + (size_t)(R0 + (hf << 3)) * DM_ + n;
#pragma unroll
  for (int r = 0; r < 8; ++r) p[r * DM_] = acc[r];
}

// ---------------------------------------------------------------------------
extern "C" void kernel_launch(void* const* d_in, const int* in_sizes, int n_in,
                              void* d_out, int out_size, void* d_ws, size_t ws_size,
                              hipStream_t stream) {
  const float* x     = (const float*)d_in[0];
  const float* Win   = (const float*)d_in[1];
  const float* convw = (const float*)d_in[2];
  const float* convb = (const float*)d_in[3];
  const float* Wxp   = (const float*)d_in[4];
  const float* Wdt   = (const float*)d_in[5];
  const float* bdt   = (const float*)d_in[6];
  const float* Alogs = (const float*)d_in[7];
  const float* Ds    = (const float*)d_in[8];
  const float* Wout  = (const float*)d_in[9];
  float* out = (float*)d_out;

  float* ws    = (float*)d_ws;
  float* xz    = ws;                                   // 16384*384  (xc | z)
  float* xconv = xz    + (size_t)16384 * NZ_;          // 16384*192
  float* dy    = xconv + (size_t)16384 * DI_;          // 65536*192 (delta, then y)
  float* BC    = dy    + (size_t)65536 * DI_;          // 65536*32  (B | C)
  f16*   WinP  = (f16*)(BC + ((size_t)65536 << 5));    // 24*3*512 f16
  f16*   WxpP  = WinP + 24 * 3 * 512;                  //  3*6*512 f16
  f16*   WoutP = WxpP +  3 * 6 * 512;                  //  6*6*512 f16
  (void)ws_size; (void)in_sizes; (void)n_in; (void)out_size;

  // pre-pack weights into WMMA B-fragment order (f16, zero-padded)
  k_pack<<<144, 256, 0, stream>>>(Win,  WinP,  DM_, NZ_, 3, 24 * 3 * 512);
  k_pack<<< 36, 256, 0, stream>>>(Wxp,  WxpP,  DI_, NX_, 6,  3 * 6 * 512);
  k_pack<<< 72, 256, 0, stream>>>(Wout, WoutP, DI_, DM_, 6,  6 * 6 * 512);

  k_inproj<<<1024, 256, 0, stream>>>(x, WinP, xz);
  k_conv  <<<16384, DI_, 0, stream>>>(xz, convw, convb, xconv);
  k_xproj <<<4096, DI_, 0, stream>>>(xconv, WxpP, Wdt, bdt, dy, BC);
  k_scan  <<<96, 32, 0, stream>>>(xconv, Alogs, dy, BC);
  k_out   <<<1024, DI_, 0, stream>>>(dy, xconv, xz, Ds, WoutP, out);
}
